// Net_19164144075490
// MI455X (gfx1250) — matrix operand order repaired and checked by reading, not compiled, not run on hardware
//
#include <hip/hip_runtime.h>
#include <math.h>

// ---------------------------------------------------------------------------
// PNA-style GNN for MI455X (gfx1250, wave32, WMMA).
// All dense GEMMs run on v_wmma_f32_16x16x32_f16 (f16 operands, f32 accum).
// LDS tiles are stored pre-swizzled into the WMMA register layout so each
// operand load is one contiguous 32B read per lane (ds_load_b128 pairs)
// instead of 16 scalar ds_load_u16 + pack moves. Tile fills use float4
// global gathers (global_load_b128) -- the real roofline limit here is the
// random gather/scatter traffic, not FLOPs.
// ---------------------------------------------------------------------------

typedef __attribute__((ext_vector_type(16))) _Float16 v16h;
typedef __attribute__((ext_vector_type(4)))  _Float16 v4h;
typedef __attribute__((ext_vector_type(8)))  float    v8f;

#define N_NODES 50000
#define N_EDGES 800000
#define N_GRAPH 128
#define L_LAYERS 4
#define T_TOW 4
#define F_FEAT 20
#define D_DIM 80
#define AVG_DEG_LOG 2.8332133f   // log(17.0)
#define EPS_STD 1e-5f
#define EPS_BN  1e-5f

#define WMMA_F32_F16(a,b,c) \
  __builtin_amdgcn_wmma_f32_16x16x32_f16(false,(a),false,(b),(short)0,(c),false,false)

// order-preserving float<->int encoding for integer atomicMin/Max
__device__ __forceinline__ int fenc(float f) {
  int i = __float_as_int(f);
  return (i >= 0) ? i : (i ^ 0x7FFFFFFF);
}
__device__ __forceinline__ float fdec(int e) {
  int i = (e >= 0) ? e : (e ^ 0x7FFFFFFF);
  return __int_as_float(i);
}

// ---- fragment-swizzled LDS tiles -----------------------------------------
// One operand fragment = 32 lanes x 16 halves, contiguous per lane.
__device__ __forceinline__ v16h ld_frag(const _Float16* buf, int slot, int lane) {
  return *(const v16h*)(buf + ((slot << 5) + lane) * 16);
}
// A-operand swizzle: element (m in 0..15, k) -> half index.
// Inverse of ISA 7.12.2: lane = g*16+m, K = slab*32 + 8g + (j&7) + ((j>>3)<<4)
__device__ __forceinline__ int a_sw_idx(int m, int k) {
  int slab = k >> 5, kk = k & 31;
  int g = (kk >> 3) & 1;
  int j = (kk & 7) + ((kk >> 4) << 3);
  return ((slab << 5) + (g << 4) + m) * 16 + j;
}
// B-operand swizzle: element (k, n); ntiles = number of 16-wide n tiles.
__device__ __forceinline__ int b_sw_idx(int k, int n, int ntiles) {
  int slab = k >> 5, kk = k & 31;
  int g = (kk >> 3) & 1, j = (kk & 7) + ((kk >> 4) << 3);
  int nt = n >> 4, nl = n & 15;
  return (((slab * ntiles + nt) << 5) + (g << 4) + nl) * 16 + j;
}
// Store 4 consecutive-k halves of an A tile (k must be a multiple of 4:
// then j is quad aligned -> single 8B LDS store).
__device__ __forceinline__ void st_a_quad(_Float16* buf, int m, int k,
                                          float x, float y, float z, float w) {
  v4h q;
  q[0] = (_Float16)x; q[1] = (_Float16)y; q[2] = (_Float16)z; q[3] = (_Float16)w;
  *(v4h*)(buf + a_sw_idx(m, k)) = q;
}

// ---------------------------------------------------------------------------
__global__ void k_init_global(float* deg, float* gsum, float* gcnt) {
  int i = blockIdx.x * blockDim.x + threadIdx.x;
  if (i < N_NODES) deg[i] = 0.f;
  if (i < N_GRAPH * D_DIM) gsum[i] = 0.f;
  if (i < N_GRAPH) gcnt[i] = 0.f;
}

__global__ void k_embed(const int* __restrict__ x, const float* __restrict__ atom_emb,
                        float* __restrict__ h) {
  int i = blockIdx.x * blockDim.x + threadIdx.x;
  if (i >= N_NODES * D_DIM) return;
  int n = i / D_DIM, d = i % D_DIM;
  float s = 0.f;
#pragma unroll
  for (int f = 0; f < 9; ++f) {
    int xv = x[(size_t)n * 9 + f];
    s += atom_emb[((size_t)f * 16 + xv) * D_DIM + d];
  }
  h[i] = s;
}

__global__ void k_deg(const int* __restrict__ edge_index, float* __restrict__ deg) {
  int e = blockIdx.x * blockDim.x + threadIdx.x;
  if (e < N_EDGES) atomicAdd(deg + edge_index[N_EDGES + e], 1.f);
}

__global__ void k_layer_init(float* agg_sum, float* agg_sq, int* agg_min, int* agg_max,
                             float* bnsum, float* bnsq) {
  int i = blockIdx.x * blockDim.x + threadIdx.x;
  if (i < N_NODES * D_DIM) {
    agg_sum[i] = 0.f; agg_sq[i] = 0.f;
    agg_min[i] = 0x7FFFFFFF;
    agg_max[i] = (int)0x80000000;
  }
  if (i < D_DIM) { bnsum[i] = 0.f; bnsq[i] = 0.f; }
}

// ---------------------------------------------------------------------------
// Edge kernel: 4 waves (one per tower), 8 tiles of 16 edges per workgroup.
// A = [xt[dst](20) | xt[src](20) | edge_attr@edge_w+edge_b (20) | 0pad] 16x64,
// GEMM1 (relu) -> GEMM2 -> 4-way atomic aggregation keyed on dst.
// ---------------------------------------------------------------------------
__global__ __launch_bounds__(128) void k_edge(
    const float* __restrict__ h, const int* __restrict__ edge_index,
    const float* __restrict__ edge_attr,
    const float* __restrict__ edge_w, const float* __restrict__ edge_b,
    const float* __restrict__ pre_w1, const float* __restrict__ pre_b1,
    const float* __restrict__ pre_w2, const float* __restrict__ pre_b2,
    float* __restrict__ agg_sum, float* __restrict__ agg_sq,
    int* __restrict__ agg_min, int* __restrict__ agg_max, int l)
{
  __shared__ __align__(32) _Float16 Bs1[T_TOW * 2048]; // 2 slabs x 2 ntiles, swizzled
  __shared__ __align__(32) _Float16 Bs2[T_TOW * 1024]; // 1 slab  x 2 ntiles
  __shared__ __align__(32) _Float16 As [T_TOW * 1024]; // 2 slabs (16x64)
  __shared__ __align__(32) _Float16 Ms [T_TOW * 512];  // 1 slab  (16x32)
  __shared__ __align__(16) float    eas[16 * F_FEAT];
  __shared__ int dsts[16], srcs[16];

  const int tid = threadIdx.x;
  const int lane = tid & 31;
  const int t = tid >> 5;               // wave == tower
  const int lanen = lane & 15;
  const int g = (lane >> 4) & 1;

  for (int i = tid; i < T_TOW * 64 * 32; i += 128) {
    int tt = i >> 11, rem = i & 2047, k = rem >> 5, n = rem & 31;
    float v = (k < 60 && n < 20) ? pre_w1[(((size_t)l * T_TOW + tt) * 60 + k) * 20 + n] : 0.f;
    Bs1[tt * 2048 + b_sw_idx(k, n, 2)] = (_Float16)v;
  }
  for (int i = tid; i < T_TOW * 32 * 32; i += 128) {
    int tt = i >> 10, rem = i & 1023, k = rem >> 5, n = rem & 31;
    float v = (k < 20 && n < 20) ? pre_w2[(((size_t)l * T_TOW + tt) * 20 + k) * 20 + n] : 0.f;
    Bs2[tt * 1024 + b_sw_idx(k, n, 2)] = (_Float16)v;
  }

  for (int it = 0; it < 8; ++it) {
    int e0 = (blockIdx.x * 8 + it) * 16;
    __syncthreads();                    // guard LDS reuse across tiles (uniform)
    if (e0 >= N_EDGES) break;

    if (tid < 16) {
      int e = min(e0 + tid, N_EDGES - 1);
      srcs[tid] = edge_index[e];
      dsts[tid] = edge_index[N_EDGES + e];
    }
    for (int i = tid; i < 16 * F_FEAT; i += 128) {   // ea = edge_attr@edge_w + edge_b
      int r = i / F_FEAT, f = i % F_FEAT;
      int e = min(e0 + r, N_EDGES - 1);
      float s = edge_b[(size_t)l * F_FEAT + f];
#pragma unroll
      for (int k = 0; k < 16; ++k)
        s += edge_attr[(size_t)e * 16 + k] * edge_w[((size_t)l * 16 + k) * F_FEAT + f];
      eas[i] = s;
    }
    __syncthreads();
    // A tiles, quad-vectorized fill (float4 gathers, 8B swizzled LDS stores)
    for (int i = tid; i < T_TOW * 16 * 16; i += 128) {
      int tt = i >> 8, rem = i & 255, r = rem >> 4, k = (rem & 15) << 2;
      float vx = 0.f, vy = 0.f, vz = 0.f, vw = 0.f;
      if (k < 20) {
        const float4 v = *(const float4*)(h + (size_t)dsts[r] * D_DIM + tt * F_FEAT + k);
        vx = v.x; vy = v.y; vz = v.z; vw = v.w;
      } else if (k < 40) {
        const float4 v = *(const float4*)(h + (size_t)srcs[r] * D_DIM + tt * F_FEAT + (k - 20));
        vx = v.x; vy = v.y; vz = v.z; vw = v.w;
      } else if (k < 60) {
        const float4 v = *(const float4*)(eas + r * F_FEAT + (k - 40));
        vx = v.x; vy = v.y; vz = v.z; vw = v.w;
      }
      st_a_quad(As + tt * 1024, r, k, vx, vy, vz, vw);
    }
    __syncthreads();

    // GEMM1: (16x64) @ (64x32), relu
    v16h a0 = ld_frag(As + t * 1024, 0, lane);
    v16h a1 = ld_frag(As + t * 1024, 1, lane);
#pragma unroll
    for (int nt = 0; nt < 2; ++nt) {
      v16h b0 = ld_frag(Bs1 + t * 2048, 0 * 2 + nt, lane);
      v16h b1 = ld_frag(Bs1 + t * 2048, 1 * 2 + nt, lane);
      v8f c = {0.f,0.f,0.f,0.f,0.f,0.f,0.f,0.f};
      c = WMMA_F32_F16(a0, b0, c);
      c = WMMA_F32_F16(a1, b1, c);
      int n = nt * 16 + lanen;
      float bias = (n < F_FEAT) ? pre_b1[((size_t)l * T_TOW + t) * F_FEAT + n] : 0.f;
#pragma unroll
      for (int r = 0; r < 8; ++r) {
        int row = r + 8 * g;
        float v = (n < F_FEAT) ? fmaxf(c[r] + bias, 0.f) : 0.f;
        Ms[t * 512 + a_sw_idx(row, n)] = (_Float16)v;   // write in A-swizzle
      }
    }
    __syncthreads();

    // GEMM2: (16x32) @ (32x32) + aggregation
    v16h a2 = ld_frag(Ms + t * 512, 0, lane);
#pragma unroll
    for (int nt = 0; nt < 2; ++nt) {
      v16h b = ld_frag(Bs2 + t * 1024, nt, lane);
      v8f c = {0.f,0.f,0.f,0.f,0.f,0.f,0.f,0.f};
      c = WMMA_F32_F16(a2, b, c);
      int n = nt * 16 + lanen;
      bool nok = (n < F_FEAT);
      float bias = nok ? pre_b2[((size_t)l * T_TOW + t) * F_FEAT + n] : 0.f;
#pragma unroll
      for (int r = 0; r < 8; ++r) {
        int row = r + 8 * g;
        int e = e0 + row;
        if (nok && e < N_EDGES) {
          float v = c[r] + bias;
          size_t addr = (size_t)dsts[row] * D_DIM + t * F_FEAT + n;
          atomicAdd(agg_sum + addr, v);
          atomicAdd(agg_sq  + addr, v * v);
          atomicMax(agg_max + addr, fenc(v));
          atomicMin(agg_min + addr, fenc(v));
        }
      }
    }
  }
}

// ---------------------------------------------------------------------------
// Node kernel: 32 nodes / iteration, 4 iterations / workgroup.
// Waves: w&1 -> node sub-block (16 nodes), w>>1 -> output n-tile.
// Per tower: A = [xt(20) | mean,mn,mx,std(80) | agg*s1(80) | agg*s2(80) | pad]
// (16x288, 9 k-steps, B streamed), then post_w2, then tower-mixing lin GEMM.
// BN stats reduced in LDS, flushed with 2 global atomics/channel.
// ---------------------------------------------------------------------------
__global__ __launch_bounds__(128) void k_node(
    const float* __restrict__ h, const float* __restrict__ deg,
    const float* __restrict__ agg_sum, const float* __restrict__ agg_sq,
    const int* __restrict__ agg_min, const int* __restrict__ agg_max,
    const float* __restrict__ post_w1, const float* __restrict__ post_b1,
    const float* __restrict__ post_w2, const float* __restrict__ post_b2,
    const float* __restrict__ lin_w, const float* __restrict__ lin_b,
    float* __restrict__ lin_out, float* __restrict__ bnsum, float* __restrict__ bnsq,
    int l)
{
  // As: 2 node-blocks x 9 slabs x 512 = 9216 halves; Blin: 3 slabs x 5 nt x 512 = 7680
  __shared__ __align__(32) unsigned char arena[9216 * 2];
  _Float16* As   = (_Float16*)arena;
  _Float16* Blin = (_Float16*)arena;
  __shared__ __align__(32) _Float16 BkS[2 * 512];       // 1 slab x 2 ntiles
  __shared__ __align__(32) _Float16 Ms [2 * 512];       // 2 node-blocks x (16x32)
  __shared__ __align__(32) _Float16 B2s[2 * 512];       // 1 slab x 2 ntiles
  __shared__ __align__(32) _Float16 Ls [2 * 3 * 512];   // 2 node-blocks x 3 slabs
  __shared__ float bnS[D_DIM], bnQ[D_DIM];
  __shared__ float nsc[32][4];   // deg(clipped), s1, s2, has_edge

  const int tid = threadIdx.x;
  const int lane = tid & 31;
  const int w = tid >> 5;
  const int lanen = lane & 15;
  const int g = (lane >> 4) & 1;
  const int nb = w & 1;               // node sub-block for GEMM1/2
  const int ntile = w >> 1;           // output n-tile for GEMM1/2
  const int rb = nb * 16;

  if (tid < D_DIM) { bnS[tid] = 0.f; bnQ[tid] = 0.f; }

  for (int itn = 0; itn < 4; ++itn) {
    int n0 = (blockIdx.x * 4 + itn) * 32;
    __syncthreads();                  // guard LDS reuse across iterations
    if (n0 >= N_NODES) break;

    if (tid < 32) {
      int node = n0 + tid;
      float dr = (node < N_NODES) ? deg[node] : 0.f;
      float dg = fmaxf(dr, 1.f);
      float ld = logf(dg + 1.f);
      nsc[tid][0] = dg;
      nsc[tid][1] = ld / AVG_DEG_LOG;
      nsc[tid][2] = AVG_DEG_LOG / ld;
      nsc[tid][3] = (dr > 0.f) ? 1.f : 0.f;
    }
    // zero-pad Ls columns 80..95 (lin K-pad)
    for (int i = tid; i < 2 * 16 * 16; i += 128) {
      int b = i >> 8, rem = i & 255, m = rem >> 4, c = 80 + (rem & 15);
      Ls[b * 1536 + a_sw_idx(m, c)] = (_Float16)0.f;
    }
    __syncthreads();

    for (int t = 0; t < T_TOW; ++t) {
      // A tile fill, quad-vectorized (32 rows x 72 quads)
      for (int i = tid; i < 32 * 72; i += 128) {
        int r = i / 72, q = i % 72;
        int k = q << 2;
        int node = min(n0 + r, N_NODES - 1);
        int bb = r >> 4, m = r & 15;
        float vx = 0.f, vy = 0.f, vz = 0.f, vw = 0.f;
        if (k < 20) {
          const float4 v = *(const float4*)(h + (size_t)node * D_DIM + t * F_FEAT + k);
          vx = v.x; vy = v.y; vz = v.z; vw = v.w;
        } else if (k < 260) {
          int qq = k - 20;
          int grp = qq / 80;          // 0: x1, 1: *s1, 2: *s2
          int idx = qq % 80;
          int st = idx / 20, f = idx % 20;
          size_t addr = (size_t)node * D_DIM + t * F_FEAT + f;
          float dg = nsc[r][0], he = nsc[r][3];
          if (st == 0) {
            const float4 s = *(const float4*)(agg_sum + addr);
            vx = s.x / dg; vy = s.y / dg; vz = s.z / dg; vw = s.w / dg;
          } else if (st == 1) {
            const int4 mi = *(const int4*)(agg_min + addr);
            vx = he * fdec(mi.x); vy = he * fdec(mi.y);
            vz = he * fdec(mi.z); vw = he * fdec(mi.w);
          } else if (st == 2) {
            const int4 ma = *(const int4*)(agg_max + addr);
            vx = he * fdec(ma.x); vy = he * fdec(ma.y);
            vz = he * fdec(ma.z); vw = he * fdec(ma.w);
          } else {
            const float4 s = *(const float4*)(agg_sum + addr);
            const float4 qs = *(const float4*)(agg_sq + addr);
            float m0 = s.x / dg, m1 = s.y / dg, m2 = s.z / dg, m3 = s.w / dg;
            vx = sqrtf(fmaxf(qs.x / dg - m0 * m0, 0.f) + EPS_STD);
            vy = sqrtf(fmaxf(qs.y / dg - m1 * m1, 0.f) + EPS_STD);
            vz = sqrtf(fmaxf(qs.z / dg - m2 * m2, 0.f) + EPS_STD);
            vw = sqrtf(fmaxf(qs.w / dg - m3 * m3, 0.f) + EPS_STD);
          }
          float sc = (grp == 0) ? 1.f : (grp == 1 ? nsc[r][1] : nsc[r][2]);
          vx *= sc; vy *= sc; vz *= sc; vw *= sc;
        }
        st_a_quad(As + bb * 4608, m, k, vx, vy, vz, vw);
      }
      for (int i = tid; i < 32 * 32; i += 128) {
        int k = i >> 5, n = i & 31;
        float v = (k < 20 && n < 20) ? post_w2[(((size_t)l * T_TOW + t) * 20 + k) * 20 + n] : 0.f;
        B2s[b_sw_idx(k, n, 2)] = (_Float16)v;
      }
      __syncthreads();

      // GEMM1: (16x288) @ (288x32), B streamed in 9 k-slices
      v8f c1 = {0.f,0.f,0.f,0.f,0.f,0.f,0.f,0.f};
      for (int ks = 0; ks < 9; ++ks) {
        for (int i = tid; i < 32 * 32; i += 128) {
          int kk = i >> 5, n = i & 31;
          int kg = ks * 32 + kk;
          float v = (kg < 260 && n < 20)
                    ? post_w1[(((size_t)l * T_TOW + t) * 260 + kg) * 20 + n] : 0.f;
          BkS[b_sw_idx(kk, n, 2)] = (_Float16)v;
        }
        __syncthreads();
        v16h a = ld_frag(As + nb * 4608, ks, lane);
        v16h b = ld_frag(BkS, ntile, lane);
        c1 = WMMA_F32_F16(a, b, c1);
        __syncthreads();
      }
      {
        int n = ntile * 16 + lanen;
        float bias = (n < F_FEAT) ? post_b1[((size_t)l * T_TOW + t) * F_FEAT + n] : 0.f;
#pragma unroll
        for (int r = 0; r < 8; ++r) {
          int m = r + 8 * g;
          float v = (n < F_FEAT) ? fmaxf(c1[r] + bias, 0.f) : 0.f;
          Ms[nb * 512 + a_sw_idx(m, n)] = (_Float16)v;
        }
      }
      __syncthreads();
      // GEMM2: (16x32) @ (32x32) -> Ls[:, t*20 .. t*20+19] (A-swizzled for lin)
      {
        v16h a = ld_frag(Ms + nb * 512, 0, lane);
        v16h b = ld_frag(B2s, ntile, lane);
        v8f c2 = {0.f,0.f,0.f,0.f,0.f,0.f,0.f,0.f};
        c2 = WMMA_F32_F16(a, b, c2);
        int n = ntile * 16 + lanen;
        if (n < F_FEAT) {
          float bias = post_b2[((size_t)l * T_TOW + t) * F_FEAT + n];
          int c_col = t * F_FEAT + n;
#pragma unroll
          for (int r = 0; r < 8; ++r) {
            int m = r + 8 * g;
            Ls[nb * 1536 + a_sw_idx(m, c_col)] = (_Float16)(c2[r] + bias);
          }
        }
      }
      __syncthreads();
    } // towers

    // lin: (32x96) @ (96x80); Blin aliases the A arena (A no longer needed)
    for (int i = tid; i < 96 * 80; i += 128) {
      int k = i / 80, n = i % 80;
      float v = (k < 80) ? lin_w[((size_t)l * 80 + k) * 80 + n] : 0.f;
      Blin[b_sw_idx(k, n, 5)] = (_Float16)v;
    }
    __syncthreads();
    for (int idx = w; idx < 10; idx += 4) {       // 2 node-blocks x 5 n-tiles
      int bb = idx / 5, nt = idx % 5;
      v8f c = {0.f,0.f,0.f,0.f,0.f,0.f,0.f,0.f};
#pragma unroll
      for (int ks = 0; ks < 3; ++ks) {
        v16h a = ld_frag(Ls + bb * 1536, ks, lane);
        v16h b = ld_frag(Blin, ks * 5 + nt, lane);
        c = WMMA_F32_F16(a, b, c);
      }
      int n = nt * 16 + lanen;                    // 0..79
      float bias = lin_b[(size_t)l * D_DIM + n];
#pragma unroll
      for (int r = 0; r < 8; ++r) {
        int row = bb * 16 + r + 8 * g;
        int node = n0 + row;
        if (node < N_NODES) {
          float v = c[r] + bias;
          lin_out[(size_t)node * D_DIM + n] = v;
          atomicAdd(&bnS[n], v);
          atomicAdd(&bnQ[n], v * v);
        }
      }
    }
  } // iterations

  __syncthreads();
  if (tid < D_DIM) {
    atomicAdd(bnsum + tid, bnS[tid]);
    atomicAdd(bnsq  + tid, bnQ[tid]);
  }
}

// ---------------------------------------------------------------------------
__global__ void k_bn_fin(const float* __restrict__ bnsum, const float* __restrict__ bnsq,
                         const float* __restrict__ bn_g, const float* __restrict__ bn_b,
                         float* __restrict__ scale, float* __restrict__ shift, int l) {
  int c = threadIdx.x;
  if (c < D_DIM) {
    float mu = bnsum[c] / (float)N_NODES;
    float var = fmaxf(bnsq[c] / (float)N_NODES - mu * mu, 0.f);
    float s = bn_g[(size_t)l * D_DIM + c] * rsqrtf(var + EPS_BN);
    scale[c] = s;
    shift[c] = bn_b[(size_t)l * D_DIM + c] - mu * s;
  }
}

__global__ void k_residual(const float* __restrict__ lin_out,
                           const float* __restrict__ scale, const float* __restrict__ shift,
                           float* __restrict__ h) {
  int i = blockIdx.x * blockDim.x + threadIdx.x;
  if (i >= N_NODES * D_DIM) return;
  int c = i % D_DIM;
  float v = lin_out[i] * scale[c] + shift[c];
  h[i] = fmaxf(v, 0.f) + h[i];
}

__global__ void k_pool(const float* __restrict__ h, const int* __restrict__ batch,
                       float* __restrict__ gsum, float* __restrict__ gcnt) {
  int i = blockIdx.x * blockDim.x + threadIdx.x;
  if (i >= N_NODES * D_DIM) return;
  int n = i / D_DIM, c = i % D_DIM;
  int b = batch[n];
  atomicAdd(gsum + (size_t)b * D_DIM + c, h[i]);
  if (c == 0) atomicAdd(gcnt + b, 1.f);
}

__global__ void k_mlp(const float* __restrict__ gsum, const float* __restrict__ gcnt,
                      const float* __restrict__ w1, const float* __restrict__ b1,
                      const float* __restrict__ w2, const float* __restrict__ b2,
                      const float* __restrict__ w3, const float* __restrict__ b3,
                      float* __restrict__ out) {
  int gph = blockIdx.x * blockDim.x + threadIdx.x;
  if (gph >= N_GRAPH) return;
  float inv = 1.f / fmaxf(gcnt[gph], 1.f);
  float gv[80];
  for (int c = 0; c < 80; ++c) gv[c] = gsum[(size_t)gph * 80 + c] * inv;
  float h1[40];
  for (int j = 0; j < 40; ++j) {
    float s = b1[j];
    for (int c = 0; c < 80; ++c) s += gv[c] * w1[(size_t)c * 40 + j];
    h1[j] = fmaxf(s, 0.f);
  }
  float h2[20];
  for (int j = 0; j < 20; ++j) {
    float s = b2[j];
    for (int c = 0; c < 40; ++c) s += h1[c] * w2[(size_t)c * 20 + j];
    h2[j] = fmaxf(s, 0.f);
  }
  float s = b3[0];
  for (int c = 0; c < 20; ++c) s += h2[c] * w3[c];
  out[gph] = s;
}

// ---------------------------------------------------------------------------
extern "C" void kernel_launch(void* const* d_in, const int* in_sizes, int n_in,
                              void* d_out, int out_size, void* d_ws, size_t ws_size,
                              hipStream_t stream) {
  (void)in_sizes; (void)n_in; (void)out_size; (void)ws_size;
  const int*   x          = (const int*)  d_in[0];
  const int*   edge_index = (const int*)  d_in[1];
  const int*   batch      = (const int*)  d_in[2];
  const float* edge_attr  = (const float*)d_in[3];
  const float* atom_emb   = (const float*)d_in[4];
  const float* edge_w     = (const float*)d_in[5];
  const float* edge_b     = (const float*)d_in[6];
  const float* pre_w1     = (const float*)d_in[7];
  const float* pre_b1     = (const float*)d_in[8];
  const float* pre_w2     = (const float*)d_in[9];
  const float* pre_b2     = (const float*)d_in[10];
  const float* post_w1    = (const float*)d_in[11];
  const float* post_b1    = (const float*)d_in[12];
  const float* post_w2    = (const float*)d_in[13];
  const float* post_b2    = (const float*)d_in[14];
  const float* lin_w      = (const float*)d_in[15];
  const float* lin_b      = (const float*)d_in[16];
  const float* bn_g       = (const float*)d_in[17];
  const float* bn_b       = (const float*)d_in[18];
  const float* mlp_w1     = (const float*)d_in[19];
  const float* mlp_b1     = (const float*)d_in[20];
  const float* mlp_w2     = (const float*)d_in[21];
  const float* mlp_b2     = (const float*)d_in[22];
  const float* mlp_w3     = (const float*)d_in[23];
  const float* mlp_b3     = (const float*)d_in[24];
  float* out = (float*)d_out;

  float* ws = (float*)d_ws;
  size_t off = 0;
  float* h       = ws + off; off += (size_t)N_NODES * D_DIM;
  float* deg     = ws + off; off += N_NODES;
  float* agg_sum = ws + off; off += (size_t)N_NODES * D_DIM;
  float* agg_sq  = ws + off; off += (size_t)N_NODES * D_DIM;
  int*   agg_min = (int*)(ws + off); off += (size_t)N_NODES * D_DIM;
  int*   agg_max = (int*)(ws + off); off += (size_t)N_NODES * D_DIM;
  float* lin_out = ws + off; off += (size_t)N_NODES * D_DIM;
  float* bnsum   = ws + off; off += D_DIM;
  float* bnsq    = ws + off; off += D_DIM;
  float* bnscale = ws + off; off += D_DIM;
  float* bnshift = ws + off; off += D_DIM;
  float* gsum    = ws + off; off += (size_t)N_GRAPH * D_DIM;
  float* gcnt    = ws + off; off += N_GRAPH;

  const int TPB = 256;
  const int gND = (N_NODES * D_DIM + TPB - 1) / TPB;

  k_init_global<<<(N_NODES + TPB - 1) / TPB, TPB, 0, stream>>>(deg, gsum, gcnt);
  k_embed<<<gND, TPB, 0, stream>>>(x, atom_emb, h);
  k_deg<<<(N_EDGES + TPB - 1) / TPB, TPB, 0, stream>>>(edge_index, deg);

  for (int l = 0; l < L_LAYERS; ++l) {
    k_layer_init<<<gND, TPB, 0, stream>>>(agg_sum, agg_sq, agg_min, agg_max, bnsum, bnsq);
    k_edge<<<(N_EDGES + 127) / 128, 128, 0, stream>>>(
        h, edge_index, edge_attr, edge_w, edge_b,
        pre_w1, pre_b1, pre_w2, pre_b2,
        agg_sum, agg_sq, agg_min, agg_max, l);
    k_node<<<(N_NODES + 127) / 128, 128, 0, stream>>>(
        h, deg, agg_sum, agg_sq, agg_min, agg_max,
        post_w1, post_b1, post_w2, post_b2, lin_w, lin_b,
        lin_out, bnsum, bnsq, l);
    k_bn_fin<<<1, 128, 0, stream>>>(bnsum, bnsq, bn_g, bn_b, bnscale, bnshift, l);
    k_residual<<<gND, TPB, 0, stream>>>(lin_out, bnscale, bnshift, h);
  }

  k_pool<<<gND, TPB, 0, stream>>>(h, batch, gsum, gcnt);
  k_mlp<<<1, N_GRAPH, 0, stream>>>(gsum, gcnt, mlp_w1, mlp_b1, mlp_w2, mlp_b2,
                                   mlp_w3, mlp_b3, out);
}
